// Conv2dLayer_14998025797815
// MI455X (gfx1250) — compile-verified, matrix operand order checked by reading, not run on hardware
//
#include <hip/hip_runtime.h>
#include <hip/hip_bf16.h>
#include <math.h>

typedef __attribute__((ext_vector_type(16))) _Float16 v16h;
typedef __attribute__((ext_vector_type(8)))  float    v8f;
typedef __attribute__((ext_vector_type(4)))  float    v4f;
typedef __attribute__((ext_vector_type(4)))  unsigned v4u;

#define IN_C    128
#define OUT_C   64
#define CDIM    512
#define HW_IN   128
#define HW_OUT  256
#define NCHUNK  36            // K = 9 taps * 128 ic = 1152, 36 chunks of 32
#define XS_STRIDE 152         // 128 ic + pad: 304B lines (16B aligned, bank spread)

// ---------------------------------------------------------------------------
// Kernel 1: per-(n, ic) modulation gains  g = 1 + tanh(c . w_affine/sqrt(512) + b)
// ---------------------------------------------------------------------------
__global__ void mod_gains_kernel(const float* __restrict__ c,
                                 const float* __restrict__ w_affine,
                                 const float* __restrict__ b_affine,
                                 float* __restrict__ g) {
    const int n  = blockIdx.x;     // 0..7
    const int ic = threadIdx.x;    // 0..127
    const float* crow = c + n * CDIM;
    const float* wrow = w_affine + ic * CDIM;
    float s = 0.f;
    for (int k = 0; k < CDIM; ++k) s += crow[k] * wrow[k];
    s = s * 0.044194173824159216f + b_affine[ic];   // 1/sqrt(512)
    g[n * IN_C + ic] = 1.0f + tanhf(s);
}

// ---------------------------------------------------------------------------
// Kernel 2: x (f32, NCHW) -> x16 (f16, NHWC) one-time convert+transpose so the
// main kernel can async-DMA ic-contiguous lines straight into LDS.
// One block per (n, h): LDS transpose of a 128(w) x 128(ic) slice.
// ---------------------------------------------------------------------------
__global__ __launch_bounds__(256)
void cvt_transpose_kernel(const float* __restrict__ x, _Float16* __restrict__ x16) {
    __shared__ _Float16 t[128 * 136];      // [w][ic], stride 136 halves (272B)
    const int h = blockIdx.x;
    const int n = blockIdx.y;
    const int tid = threadIdx.x;
    // load: coalesced along w, 4 floats per thread-iter
    for (int i = tid; i < 4096; i += 256) {
        const int ic = i >> 5;
        const int w4 = (i & 31) * 4;
        const v4f v = *(const v4f*)&x[(((long long)n * IN_C + ic) * HW_IN + h) * HW_IN + w4];
        t[(w4 + 0) * 136 + ic] = (_Float16)v.x;
        t[(w4 + 1) * 136 + ic] = (_Float16)v.y;
        t[(w4 + 2) * 136 + ic] = (_Float16)v.z;
        t[(w4 + 3) * 136 + ic] = (_Float16)v.w;
    }
    __syncthreads();
    // store: coalesced along ic, 8 halves (16B) per thread-iter
    for (int i = tid; i < 2048; i += 256) {
        const int w   = i >> 4;
        const int icb = (i & 15) * 8;
        const v4u val = *(const v4u*)&t[w * 136 + icb];
        *(v4u*)&x16[(((long long)n * HW_IN + h) * HW_IN + w) * IN_C + icb] = val;
    }
}

// ---------------------------------------------------------------------------
// Kernel 3: build modulated, parity-combined weights directly in the WMMA
// B-fragment layout:  [n][ph][pw][kc:36][ocTile:4][lane:32][half:16]  (f16)
// We = sum_{kh,kw} w[oc,ic,2-kh,2-kw]/sqrt(1152) * k4[2tH+(1-ph)-kh][2tW+(1-pw)-kw]
// with k4 = up^2 * outer(f1,f1), then scaled by gain g[n][ic].
// ---------------------------------------------------------------------------
__global__ void build_wmod_kernel(const float* __restrict__ weight,
                                  const float* __restrict__ g,
                                  _Float16* __restrict__ wmod) {
    long long idx = (long long)blockIdx.x * blockDim.x + threadIdx.x;
    const int h    = (int)(idx & 15);
    const int lane = (int)((idx >> 4) & 31);
    const int ot   = (int)((idx >> 9) & 3);
    long long t = idx >> 11;
    const int kc = (int)(t % NCHUNK); t /= NCHUNK;
    const int pw = (int)(t & 1);
    const int ph = (int)((t >> 1) & 1);
    const int n  = (int)(t >> 2);

    const int G  = lane >> 4;
    const int N  = lane & 15;
    const int kk = ((h < 8) ? h : (h + 8)) + 8 * G;   // invert fragment layout
    const int K  = kc * 32 + kk;
    const int tap = K >> 7;          // tap-major K
    const int ic  = K & 127;
    const int oc  = ot * 16 + N;
    const int tiH = tap / 3, tiW = tap % 3;
    const int sH  = 2 * tiH + (1 - ph);
    const int sW  = 2 * tiW + (1 - pw);

    const float fv[4] = {1.f, 3.f, 3.f, 1.f};
    float sum = 0.f;
    #pragma unroll
    for (int kh = 0; kh < 3; ++kh) {
        const int iH = sH - kh;
        if (iH < 0 || iH > 3) continue;
        #pragma unroll
        for (int kw = 0; kw < 3; ++kw) {
            const int iW = sW - kw;
            if (iW < 0 || iW > 3) continue;
            const float wf = weight[((oc * IN_C + ic) * 3 + (2 - kh)) * 3 + (2 - kw)];
            sum += wf * fv[iH] * fv[iW];
        }
    }
    sum *= 0.029462782549439483f * (1.0f / 16.0f);   // 1/sqrt(1152) * up^2/64
    wmod[idx] = (_Float16)(sum * g[n * IN_C + ic]);
}

// ---------------------------------------------------------------------------
// Kernel 4: fused implicit GEMM.  Block = 128 threads (4 waves) computes a
// 32-col strip of FOUR same-parity output rows (oh = rg*8+ph+2j, j=0..3):
// B fragments (even/odd col parity) are loaded once per K-chunk and reused
// across 4 row accumulators (8 WMMAs per chunk per wave).
// Input halo (6 rows x 18 cols x 128 ic, f16) staged via async LDS DMA.
// ---------------------------------------------------------------------------
__global__ __launch_bounds__(128)
void fused_upconv_kernel(const _Float16* __restrict__ x16,
                         const _Float16* __restrict__ wmod,
                         const float* __restrict__ bias,
                         float* __restrict__ out) {
    __shared__ _Float16 xs[108 * XS_STRIDE];   // [seg = r*18+c : 108][ic:128+pad]

    const int owt = blockIdx.x;        // 0..7  -> output cols [32*owt, 32*owt+32)
    const int rb  = blockIdx.y;        // 0..63 -> parity + row group
    const int n   = blockIdx.z;        // 0..7
    const int ph  = rb & 1;
    const int rg  = rb >> 1;           // 0..31
    const int tid = threadIdx.x;
    const int cwb  = owt * 16;         // base input col
    const int ihb0 = rg * 4;           // base input row of first output row

    // ---- async-DMA stage: 6 rows x 18 cols of 128-ic f16 lines ----
    {
        const int sub = tid & 15;                       // 16B granule within line
        for (int i = tid; i < 1728; i += 128) {         // 108 segs * 16 granules
            const int seg = i >> 4;
            const int r = seg / 18, cc = seg - r * 18;
            const int hh = ihb0 - 1 + r;
            const int ww = cwb - 1 + cc;
            const int lidx = seg * XS_STRIDE + sub * 8;
            if ((unsigned)hh < (unsigned)HW_IN && (unsigned)ww < (unsigned)HW_IN) {
                const _Float16* gp =
                    x16 + ((((long long)n * HW_IN + hh) * HW_IN + ww) * IN_C + sub * 8);
                const unsigned lds_off = (unsigned)(unsigned long long)(void*)&xs[lidx];
                asm volatile("global_load_async_to_lds_b128 %0, %1, off"
                             :: "v"(lds_off), "v"(gp) : "memory");
            } else {
                *(v4u*)&xs[lidx] = (v4u){0u, 0u, 0u, 0u};   // zero halo
            }
        }
        asm volatile("s_wait_asynccnt 0x0" ::: "memory");
    }
    __syncthreads();

    const int lane = tid & 31;
    const int ot   = tid >> 5;          // wave id -> oc tile (16 oc)
    const int mrow = lane & 15;         // A-fragment M index
    const int G    = lane >> 4;

    const _Float16* wE = wmod + (long long)((n * 2 + ph) * 2 + 0) * NCHUNK * 2048;
    const _Float16* wO = wmod + (long long)((n * 2 + ph) * 2 + 1) * NCHUNK * 2048;

    v8f accE[4] = {}; v8f accO[4] = {};
    union AF { v16h h; unsigned u[8]; };

    for (int kc = 0; kc < NCHUNK; ++kc) {
        const int tap    = kc >> 2;           // 4 chunks of 32 ic per tap
        const int th     = tap / 3;
        const int tw     = tap % 3;
        const int icBase = (kc & 3) * 32;

        // B fragments: one pair of coalesced 32B/lane loads per chunk,
        // reused by all 4 row accumulators (L2 resident).
        const long long boff = ((long long)kc * 4 + ot) * 512 + lane * 16;
        const v16h bE = *(const v16h*)(wE + boff);
        const v16h bO = *(const v16h*)(wO + boff);

        #pragma unroll
        for (int j = 0; j < 4; ++j) {         // 4 same-parity output rows
            const _Float16* p0 =
                &xs[((j + th) * 18 + (mrow + tw)) * XS_STRIDE + icBase + 8 * G];
            AF a;
            a.u[0] = *(const unsigned*)(p0 + 0);
            a.u[1] = *(const unsigned*)(p0 + 2);
            a.u[2] = *(const unsigned*)(p0 + 4);
            a.u[3] = *(const unsigned*)(p0 + 6);
            a.u[4] = *(const unsigned*)(p0 + 16);
            a.u[5] = *(const unsigned*)(p0 + 18);
            a.u[6] = *(const unsigned*)(p0 + 20);
            a.u[7] = *(const unsigned*)(p0 + 22);
            accE[j] = __builtin_amdgcn_wmma_f32_16x16x32_f16(
                          false, a.h, false, bE, (short)0, accE[j], false, false);
            accO[j] = __builtin_amdgcn_wmma_f32_16x16x32_f16(
                          false, a.h, false, bO, (short)0, accO[j], false, false);
        }
    }

    // ---- epilogue: bias + leaky_relu(0.2)*sqrt(2) + clamp ----
    const int   oc   = ot * 16 + (lane & 15);     // D: N = lane%16
    const float b    = bias[oc];
    const float gain = 1.4142135623730951f;
    #pragma unroll
    for (int j = 0; j < 4; ++j) {
        const int oh = rg * 8 + ph + 2 * j;
        float* obase = out + (((long long)n * OUT_C + oc) * HW_OUT + oh) * HW_OUT + owt * 32;
        #pragma unroll
        for (int v = 0; v < 8; ++v) {
            const int mm = 8 * G + v;             // D: M = vgpr + 8*(lane/16)
            float zE = accE[j][v] + b;
            zE = (zE > 0.f ? zE : 0.2f * zE) * gain;
            zE = fminf(fmaxf(zE, -256.f), 256.f);
            float zO = accO[j][v] + b;
            zO = (zO > 0.f ? zO : 0.2f * zO) * gain;
            zO = fminf(fmaxf(zO, -256.f), 256.f);
            obase[2 * mm]     = zE;               // even-parity column
            obase[2 * mm + 1] = zO;               // odd-parity column
        }
    }
}

// ---------------------------------------------------------------------------
extern "C" void kernel_launch(void* const* d_in, const int* in_sizes, int n_in,
                              void* d_out, int out_size, void* d_ws, size_t ws_size,
                              hipStream_t stream) {
    const float* x        = (const float*)d_in[0];
    const float* c        = (const float*)d_in[1];
    const float* weight   = (const float*)d_in[2];
    const float* bias     = (const float*)d_in[3];
    const float* w_affine = (const float*)d_in[4];
    const float* b_affine = (const float*)d_in[5];
    float* out = (float*)d_out;

    // workspace: [0,4KB) gains | [4KB, +9.44MB) wmod f16 | then x16 f16 (33.55MB)
    float*    g    = (float*)d_ws;
    _Float16* wmod = (_Float16*)((char*)d_ws + 4096);
    _Float16* x16  = (_Float16*)((char*)d_ws + 4096 + 9437184);

    mod_gains_kernel<<<dim3(8), dim3(128), 0, stream>>>(c, w_affine, b_affine, g);

    cvt_transpose_kernel<<<dim3(128, 8), dim3(256), 0, stream>>>(x, x16);

    // 8n * 2ph * 2pw * 36kc * 4ot * 32lane * 16half = 2,359,296 elements; grid is 2x that
    build_wmod_kernel<<<dim3(18432), dim3(256), 0, stream>>>(weight, g, wmod);

    // 8 col-tiles x (2 parities * 32 row-groups) x 8 batch, 128 threads
    fused_upconv_kernel<<<dim3(8, 64, 8), dim3(128), 0, stream>>>(x16, wmod, bias, out);
}
